// LSTMLM_64905545777783
// MI455X (gfx1250) — compile-verified, matrix operand order checked by reading, not run on hardware
//
#include <hip/hip_runtime.h>
#include <math.h>

// ---------- problem constants ----------
constexpr int V_ = 10000, E_ = 512, H_ = 1024, T_ = 128, B_ = 64;
constexpr int G_ = 4 * H_;          // 4096 (gate width)
constexpr int R_ = T_ * B_;         // 8192 (rows = T*B)

// ---------- vector types for WMMA ----------
typedef __attribute__((ext_vector_type(16))) __bf16 v16bf;
typedef __attribute__((ext_vector_type(8)))  __bf16 v8bf;
typedef __attribute__((ext_vector_type(8)))  float  v8f;

// ---------------------------------------------------------------------------
// Fragment helpers (CDNA5 wave32 WMMA 16x16x32 bf16 layouts, ISA 7.12.2)
// Per-lane base = row (row0 + lane&15), element offset (lane>=16 ? 8 : 0).
// frag_at(base, k0): two 16B loads at compile-time-constant byte offsets.
// ---------------------------------------------------------------------------
__device__ __forceinline__ v16bf frag_at(const __bf16* base, int k0) {
    v8bf lo = *(const v8bf*)(base + k0);
    v8bf hi = *(const v8bf*)(base + k0 + 16);
    return __builtin_shufflevector(lo, hi, 0, 1, 2, 3, 4, 5, 6, 7,
                                           8, 9, 10, 11, 12, 13, 14, 15);
}

// C/D tile: VGPR r -> row m0 + r + (lane>=16 ? 8 : 0), col n0 + (lane&15)
__device__ __forceinline__ v8f load_ctile_f32(const float* C, int ldc,
                                              int m0, int n0, int lane) {
    const int half = (lane >> 4) & 1;
    const int col  = n0 + (lane & 15);
    v8f acc;
#pragma unroll
    for (int r = 0; r < 8; ++r)
        acc[r] = C[(size_t)(m0 + r + half * 8) * ldc + col];
    return acc;
}

template <bool NT>
__device__ __forceinline__ void store_ctile_f32(float* C, int ldc, int m0, int n0,
                                                int lane, v8f acc, float bias) {
    const int half = (lane >> 4) & 1;
    const int col  = n0 + (lane & 15);
#pragma unroll
    for (int r = 0; r < 8; ++r) {
        float* p = C + (size_t)(m0 + r + half * 8) * ldc + col;
        float  v = acc[r] + bias;
        if (NT)
            __builtin_nontemporal_store(v, p);
        else
            *p = v;
    }
}

// ---------------------------------------------------------------------------
// Register-blocked GEMM core with hoisted per-lane base pointers and a fully
// unrolled K loop (all load offsets become immediates).
// acc[BM][BN] += A(m0.., K) * W(n0.., K)^T
// ---------------------------------------------------------------------------
template <int BM, int BN, int K>
__device__ __forceinline__ void gemm_block(const __bf16* __restrict__ A, int lda,
                                           const __bf16* __restrict__ W, int ldw,
                                           int m0, int n0, int lane,
                                           v8f (&acc)[BM][BN]) {
    const int half = (lane >> 4) & 1;
    const int r    = lane & 15;
    const __bf16* abase[BM];
#pragma unroll
    for (int i = 0; i < BM; ++i)
        abase[i] = A + (size_t)(m0 + 16 * i + r) * lda + half * 8;
    const __bf16* wbase[BN];
#pragma unroll
    for (int j = 0; j < BN; ++j)
        wbase[j] = W + (size_t)(n0 + 16 * j + r) * ldw + half * 8;

#pragma unroll
    for (int k0 = 0; k0 < K; k0 += 32) {
        if (k0 + 32 < K) {
            __builtin_prefetch(abase[0] + k0 + 32, 0, 1);
            __builtin_prefetch(wbase[0] + k0 + 32, 0, 1);
        }
        v16bf af[BM];
#pragma unroll
        for (int i = 0; i < BM; ++i)
            af[i] = frag_at(abase[i], k0);
#pragma unroll
        for (int j = 0; j < BN; ++j) {
            v16bf b = frag_at(wbase[j], k0);
#pragma unroll
            for (int i = 0; i < BM; ++i)
                acc[i][j] = __builtin_amdgcn_wmma_f32_16x16x32_bf16(
                    /*neg_a=*/false, af[i], /*neg_b=*/false, b,
                    /*c_mod=*/(short)0, acc[i][j],
                    /*reuse_a=*/false, /*reuse_b=*/false);
        }
    }
}

// ---------------------------------------------------------------------------
// Small utility kernels
// ---------------------------------------------------------------------------
__global__ void k_f32_to_bf16(const float* __restrict__ src,
                              __bf16* __restrict__ dst, int n) {
    for (int i = blockIdx.x * blockDim.x + threadIdx.x; i < n;
         i += gridDim.x * blockDim.x)
        dst[i] = (__bf16)src[i];
}

// gather embeddings for all (t,b) into bf16 activation matrix (R_ x E_)
__global__ void k_embed(const int* __restrict__ x, const float* __restrict__ emb,
                        __bf16* __restrict__ out) {
    const int n = R_ * E_;
    for (int i = blockIdx.x * blockDim.x + threadIdx.x; i < n;
         i += gridDim.x * blockDim.x) {
        int e   = i % E_;
        int row = i / E_;          // row = t*B + b, matches x layout (T,B)
        int tok = x[row];
        out[i] = (__bf16)emb[(size_t)tok * E_ + e];
    }
}

__global__ void k_init_state(const float* __restrict__ h0,
                             const float* __restrict__ c0,
                             __bf16* __restrict__ h_b,
                             float* __restrict__ c_f32) {
    const int n = B_ * H_;
    for (int i = blockIdx.x * blockDim.x + threadIdx.x; i < n;
         i += gridDim.x * blockDim.x) {
        h_b[i]   = (__bf16)h0[i];
        c_f32[i] = c0[i];
    }
}

// ---------------------------------------------------------------------------
// GEMM 1: xi[R_ x G_] = A_emb[R_ x E_] @ W_ih^T + (b_ih + b_hh)
// 4x4 register blocking (64x64 macro-tile); low-occupancy / high-VGPR budget.
// ---------------------------------------------------------------------------
__global__ void __launch_bounds__(256, 1)
k_gemm_xi(const __bf16* __restrict__ A,
          const __bf16* __restrict__ Wih,
          const float* __restrict__ b_ih,
          const float* __restrict__ b_hh,
          float* __restrict__ xi) {
    const int lane    = threadIdx.x & 31;
    const int gid     = blockIdx.x * 8 + (threadIdx.x >> 5);
    const int nblocks = G_ / 64;                // 64
    const int m0 = (gid / nblocks) * 64;
    const int n0 = (gid % nblocks) * 64;
    v8f acc[4][4] = {};
    gemm_block<4, 4, E_>(A, E_, Wih, E_, m0, n0, lane, acc);
#pragma unroll
    for (int j = 0; j < 4; ++j) {
        const int col = n0 + 16 * j + (lane & 15);
        const float bias = b_ih[col] + b_hh[col];
#pragma unroll
        for (int i = 0; i < 4; ++i)
            store_ctile_f32<false>(xi, G_, m0 + 16 * i, n0 + 16 * j, lane,
                                   acc[i][j], bias);
    }
}

// ---------------------------------------------------------------------------
// Recurrence GEMM: gates[B_ x G_] = xi_t + h_b[B_ x H_] @ W_hh^T
// 1 tile / wave: keeps 1024 waves in flight for the latency-critical chain.
// ---------------------------------------------------------------------------
__global__ void k_step_gemm(const __bf16* __restrict__ h_b,
                            const __bf16* __restrict__ Whh,
                            const float* __restrict__ xi_t,
                            float* __restrict__ gates) {
    const int lane   = threadIdx.x & 31;
    const int gid    = blockIdx.x * 8 + (threadIdx.x >> 5);
    const int mtiles = B_ / 16;                 // 4
    const int m0 = (gid % mtiles) * 16;
    const int n0 = (gid / mtiles) * 16;
    v8f acc[1][1];
    acc[0][0] = load_ctile_f32(xi_t, G_, m0, n0, lane);
    gemm_block<1, 1, H_>(h_b, H_, Whh, H_, m0, n0, lane, acc);
    store_ctile_f32<false>(gates, G_, m0, n0, lane, acc[0][0], 0.0f);
}

// ---------------------------------------------------------------------------
// Pointwise LSTM cell update (torch gate order i,f,g,o)
// ---------------------------------------------------------------------------
__global__ void k_lstm_point(const float* __restrict__ gates,
                             float* __restrict__ c_f32,
                             __bf16* __restrict__ h_b,
                             __bf16* __restrict__ hs_t) {
    const int n = B_ * H_;
    for (int idx = blockIdx.x * blockDim.x + threadIdx.x; idx < n;
         idx += gridDim.x * blockDim.x) {
        int b = idx / H_;
        int j = idx % H_;
        const float* g4 = gates + (size_t)b * G_;
        float ig = 1.0f / (1.0f + expf(-g4[j]));
        float fg = 1.0f / (1.0f + expf(-g4[H_ + j]));
        float gg = tanhf(g4[2 * H_ + j]);
        float og = 1.0f / (1.0f + expf(-g4[3 * H_ + j]));
        float c  = fg * c_f32[idx] + ig * gg;
        float h  = og * tanhf(c);
        c_f32[idx] = c;
        __bf16 hb = (__bf16)h;
        h_b[idx]  = hb;   // feeds next step's GEMM
        hs_t[idx] = hb;   // feeds projection GEMM
    }
}

// ---------------------------------------------------------------------------
// GEMM 2: logits[R_ x V_] = hs_b[R_ x H_] @ W_proj^T + b_proj
// 4x5 register blocking (64x80 macro-tile, 625 N-tiles = 125 blocks of 5).
// Non-temporal stores: logits are a 328 MB write-once stream; keep L2 for W.
// ---------------------------------------------------------------------------
__global__ void __launch_bounds__(256, 1)
k_gemm_proj(const __bf16* __restrict__ hs_b,
            const __bf16* __restrict__ Wp,
            const float* __restrict__ b_proj,
            float* __restrict__ logits) {
    const int lane    = threadIdx.x & 31;
    const int gid     = blockIdx.x * 8 + (threadIdx.x >> 5);
    const int nblocks = V_ / 80;                // 125
    const int m0 = (gid / nblocks) * 64;
    const int n0 = (gid % nblocks) * 80;
    v8f acc[4][5] = {};
    gemm_block<4, 5, H_>(hs_b, H_, Wp, H_, m0, n0, lane, acc);
#pragma unroll
    for (int j = 0; j < 5; ++j) {
        const float bias = b_proj[n0 + 16 * j + (lane & 15)];
#pragma unroll
        for (int i = 0; i < 4; ++i)
            store_ctile_f32<true>(logits, V_, m0 + 16 * i, n0 + 16 * j, lane,
                                  acc[i][j], bias);
    }
}

// ---------------------------------------------------------------------------
// argmax over V per (t,b) row; y_preds is (B,T): out[b*T + t]
// first-max tie-break like jnp.argmax; streaming (non-temporal) reads.
// ---------------------------------------------------------------------------
__global__ void k_argmax(const float* __restrict__ logits,
                         float* __restrict__ ypred) {
    __shared__ float sv[256];
    __shared__ int   si[256];
    const int row = blockIdx.x;           // row = t*B + b
    const float* p = logits + (size_t)row * V_;
    float best = -INFINITY;
    int   bidx = 0;
    for (int v = threadIdx.x; v < V_; v += blockDim.x) {
        float val = __builtin_nontemporal_load(p + v);
        if (val > best) { best = val; bidx = v; }
    }
    sv[threadIdx.x] = best;
    si[threadIdx.x] = bidx;
    __syncthreads();
    for (int off = 128; off > 0; off >>= 1) {
        if (threadIdx.x < off) {
            float v2 = sv[threadIdx.x + off];
            int   i2 = si[threadIdx.x + off];
            if (v2 > sv[threadIdx.x] ||
                (v2 == sv[threadIdx.x] && i2 < si[threadIdx.x])) {
                sv[threadIdx.x] = v2;
                si[threadIdx.x] = i2;
            }
        }
        __syncthreads();
    }
    if (threadIdx.x == 0) {
        int t = row / B_, b = row % B_;
        ypred[(size_t)b * T_ + t] = (float)si[0];
    }
}

// ---------------------------------------------------------------------------
extern "C" void kernel_launch(void* const* d_in, const int* in_sizes, int n_in,
                              void* d_out, int out_size, void* d_ws, size_t ws_size,
                              hipStream_t stream) {
    (void)in_sizes; (void)n_in; (void)out_size; (void)ws_size;
    const int*   x      = (const int*)d_in[0];
    // d_in[1] = lens (unused by the reference computation)
    const float* emb    = (const float*)d_in[2];
    const float* W_ih   = (const float*)d_in[3];
    const float* W_hh   = (const float*)d_in[4];
    const float* b_ih   = (const float*)d_in[5];
    const float* b_hh   = (const float*)d_in[6];
    const float* W_proj = (const float*)d_in[7];
    const float* b_proj = (const float*)d_in[8];
    const float* h0     = (const float*)d_in[9];
    const float* c0     = (const float*)d_in[10];
    float* out = (float*)d_out;

    // ---- workspace carve (256B aligned chunks) ----
    char* p = (char*)d_ws;
    auto carve = [&](size_t bytes) {
        void* r = (void*)p;
        p += (bytes + 255) & ~(size_t)255;
        return r;
    };
    __bf16* wih_b   = (__bf16*)carve((size_t)G_ * E_ * 2);   //  4 MB
    __bf16* whh_b   = (__bf16*)carve((size_t)G_ * H_ * 2);   //  8 MB
    __bf16* wproj_b = (__bf16*)carve((size_t)V_ * H_ * 2);   // 20 MB
    __bf16* embA    = (__bf16*)carve((size_t)R_ * E_ * 2);   //  8 MB
    float*  xi      = (float*) carve((size_t)R_ * G_ * 4);   // 128 MB
    __bf16* hs_b    = (__bf16*)carve((size_t)R_ * H_ * 2);   // 16 MB
    float*  c_f32   = (float*) carve((size_t)B_ * H_ * 4);
    __bf16* h_b     = (__bf16*)carve((size_t)B_ * H_ * 2);
    float*  gates   = (float*) carve((size_t)B_ * G_ * 4);

    // ---- 1) precision conversion + embedding gather ----
    k_f32_to_bf16<<<2048, 256, 0, stream>>>(W_ih,   wih_b,   G_ * E_);
    k_f32_to_bf16<<<2048, 256, 0, stream>>>(W_hh,   whh_b,   G_ * H_);
    k_f32_to_bf16<<<2048, 256, 0, stream>>>(W_proj, wproj_b, V_ * H_);
    k_embed<<<2048, 256, 0, stream>>>(x, emb, embA);
    k_init_state<<<(B_ * H_) / 256, 256, 0, stream>>>(h0, c0, h_b, c_f32);

    // ---- 2) hoisted input GEMM: xi = embA @ W_ih^T + bias ----
    // 64x64 macro-tiles: (8192/64)*(4096/64) = 8192 waves -> 1024 blocks
    k_gemm_xi<<<(R_ / 64) * (G_ / 64) / 8, 256, 0, stream>>>(embA, wih_b,
                                                             b_ih, b_hh, xi);

    // ---- 3) sequential recurrence ----
    // step GEMM: (B_/16)*(G_/16) = 4*256 = 1024 waves -> 128 blocks
    for (int t = 0; t < T_; ++t) {
        k_step_gemm<<<(B_ / 16) * (G_ / 16) / 8, 256, 0, stream>>>(
            h_b, whh_b, xi + (size_t)t * B_ * G_, gates);
        k_lstm_point<<<(B_ * H_) / 256, 256, 0, stream>>>(
            gates, c_f32, h_b, hs_b + (size_t)t * B_ * H_);
    }

    // ---- 4) LM head: logits = hs @ W_proj^T + b_proj -> d_out ----
    // 64x80 macro-tiles: (8192/64)*(10000/80) = 16000 waves -> 2000 blocks
    k_gemm_proj<<<(R_ / 64) * (V_ / 80) / 8, 256, 0, stream>>>(
        hs_b, wproj_b, b_proj, out);

    // ---- 5) argmax -> y_preds (B,T) appended after logits ----
    k_argmax<<<R_, 256, 0, stream>>>(out, out + (size_t)R_ * V_);
}
